// SelfAttentionHead_65506841199230
// MI455X (gfx1250) — matrix operand order repaired and checked by reading, not compiled, not run on hardware
//
#include <hip/hip_runtime.h>

// Problem constants from the reference
constexpr int Bb = 8;
constexpr int Tt = 2048;
constexpr int Cc = 1024;
constexpr int Hh = 64;

typedef __attribute__((ext_vector_type(16))) __bf16 v16bf;
typedef __attribute__((ext_vector_type(8)))  float  v8f;

// Fragment container: 32 bytes = one WMMA bf16 A/B operand per lane.
union Frag {
    v16bf v;
    uint4 u[2];
    __bf16 h[16];
    unsigned short s[16];
};

__device__ __forceinline__ unsigned short bf_bits(float f) {
    return __builtin_bit_cast(unsigned short, (__bf16)f);   // native cvt, RNE
}

// ---------------------------------------------------------------------------
// Kernel 1: repack Wq/Wk/Wv (fp32 [C,H]) into bf16 B-fragment layout:
//   packed[mat][((kc*4 + ng)*32 + lane)*16 + e]  where
//   k = kc*32 + (lane>=16 ? 16 : 0) + e,  n = ng*16 + (lane&15)
// so a lane's B operand is one contiguous 32-byte chunk.
// ---------------------------------------------------------------------------
__global__ void pack_w_kernel(const float* __restrict__ Wq,
                              const float* __restrict__ Wk,
                              const float* __restrict__ Wv,
                              unsigned short* __restrict__ Wp) {
    int tid  = blockIdx.x * 256 + threadIdx.x;     // 3 * 65536 total
    int mat  = tid >> 16;                          // 0=Q 1=K 2=V
    int idx  = tid & 65535;
    int e    = idx & 15;
    int lane = (idx >> 4) & 31;
    int ng   = (idx >> 9) & 3;
    int kc   = idx >> 11;
    int k = kc * 32 + ((lane >> 4) << 4) + e;
    int n = ng * 16 + (lane & 15);
    const float* W = (mat == 0) ? Wq : (mat == 1) ? Wk : Wv;
    Wp[tid] = bf_bits(W[k * Hh + n]);
}

// ---------------------------------------------------------------------------
// Kernel 2: fused projections. One wave = one (batch, 16-row tile); X rows are
// loaded and converted to a bf16 A-fragment ONCE per K-step and fed to 12
// WMMAs (Q,K,V x 4 column groups). Q,K stored row-major bf16 [B,T,H];
// V stored transposed bf16 [B,H,T] so the flash P*V B-fragments are
// contiguous.
// ---------------------------------------------------------------------------
__global__ __launch_bounds__(128)
void proj_kernel(const float* __restrict__ X,
                 const unsigned short* __restrict__ Wp,
                 unsigned short* __restrict__ Qbf,
                 unsigned short* __restrict__ Kbf,
                 unsigned short* __restrict__ Vt) {
    const int lane = threadIdx.x & 31;
    const int hf   = lane >> 4;        // half-wave select
    const int lr   = lane & 15;
    const int wid  = blockIdx.x * 4 + (threadIdx.x >> 5);   // [0, 1024)
    const int b    = wid >> 7;
    const int t0   = (wid & 127) << 4;

    const float* xrow = X + (size_t)((b * Tt + t0 + lr)) * Cc;

    v8f acc[3][4];
    #pragma unroll
    for (int m = 0; m < 3; ++m)
        #pragma unroll
        for (int ng = 0; ng < 4; ++ng)
            acc[m][ng] = v8f{};

    for (int kc = 0; kc < Cc / 32; ++kc) {
        const int k0 = kc * 32;
        // A fragment (contiguous k order, same as packed B): lane<16 -> K {0..15}, lane>=16 -> K {16..31} of its row
        Frag a;
        const float* pA = xrow + k0 + hf * 16;
        float4 x0 = *reinterpret_cast<const float4*>(pA);
        float4 x1 = *reinterpret_cast<const float4*>(pA + 4);
        const float* pB = xrow + k0 + hf * 16 + 8;
        float4 x2 = *reinterpret_cast<const float4*>(pB);
        float4 x3 = *reinterpret_cast<const float4*>(pB + 4);
        a.h[0]  = (__bf16)x0.x; a.h[1]  = (__bf16)x0.y; a.h[2]  = (__bf16)x0.z; a.h[3]  = (__bf16)x0.w;
        a.h[4]  = (__bf16)x1.x; a.h[5]  = (__bf16)x1.y; a.h[6]  = (__bf16)x1.z; a.h[7]  = (__bf16)x1.w;
        a.h[8]  = (__bf16)x2.x; a.h[9]  = (__bf16)x2.y; a.h[10] = (__bf16)x2.z; a.h[11] = (__bf16)x2.w;
        a.h[12] = (__bf16)x3.x; a.h[13] = (__bf16)x3.y; a.h[14] = (__bf16)x3.z; a.h[15] = (__bf16)x3.w;

        #pragma unroll
        for (int m = 0; m < 3; ++m) {
            #pragma unroll
            for (int ng = 0; ng < 4; ++ng) {
                const unsigned short* wb =
                    Wp + m * 65536 + ((kc * 4 + ng) * 32 + lane) * 16;
                Frag bf;
                bf.u[0] = *reinterpret_cast<const uint4*>(wb);
                bf.u[1] = *reinterpret_cast<const uint4*>(wb + 8);
                acc[m][ng] = __builtin_amdgcn_wmma_f32_16x16x32_bf16(
                    false, a.v, false, bf.v, (short)0, acc[m][ng], false, false);
            }
        }
    }

    // D layout: lane<16 -> (M=r, N=lr); lane>=16 -> (M=r+8, N=lr)
    #pragma unroll
    for (int ng = 0; ng < 4; ++ng)
        #pragma unroll
        for (int r = 0; r < 8; ++r) {
            const int row = t0 + r + hf * 8;
            Qbf[(b * Tt + row) * Hh + ng * 16 + lr] = bf_bits(acc[0][ng][r]);
            Kbf[(b * Tt + row) * Hh + ng * 16 + lr] = bf_bits(acc[1][ng][r]);
            Vt[(b * Hh + ng * 16 + lr) * Tt + row]  = bf_bits(acc[2][ng][r]);
        }
}

// ---------------------------------------------------------------------------
// Kernel 3: flash attention. One wave (32-thread block) per 16-query tile.
// Per 32-key block: 4 WMMAs for S = Q K^T, online softmax (shfl_xor row
// reductions within 16-lane halves), P transposed D->A layout through LDS,
// 4 WMMAs for O += P V. Causal mask folded into the score select.
// ---------------------------------------------------------------------------
__global__ __launch_bounds__(32)
void flash_kernel(const unsigned short* __restrict__ Qbf,
                  const unsigned short* __restrict__ Kbf,
                  const unsigned short* __restrict__ Vt,
                  float* __restrict__ out) {
    __shared__ __attribute__((aligned(16))) unsigned short Ptile[16 * 32];

    const int lane = threadIdx.x;
    const int hf   = lane >> 4;
    const int lr   = lane & 15;
    const int bid  = blockIdx.x;           // [0, 1024)
    const int b    = bid >> 7;
    const int t0   = (bid & 127) << 4;
    const float scale = 0.125f;            // 1/sqrt(64)

    // Q A-fragments for head-dim steps 0..31 and 32..63 (reused all blocks)
    Frag qa[2];
    const unsigned short* qrow = Qbf + (b * Tt + t0 + lr) * Hh;
    #pragma unroll
    for (int k = 0; k < 2; ++k) {
        qa[k].u[0] = *reinterpret_cast<const uint4*>(qrow + k * 32 + hf * 16);
        qa[k].u[1] = *reinterpret_cast<const uint4*>(qrow + k * 32 + hf * 16 + 8);
    }

    v8f o[4] = {v8f{}, v8f{}, v8f{}, v8f{}};
    float mi[8], li[8];
    #pragma unroll
    for (int r = 0; r < 8; ++r) { mi[r] = -1e30f; li[r] = 0.0f; }

    const int nblk = (t0 + 47) >> 5;       // 32-key blocks covering keys <= t0+15

    for (int blk = 0; blk < nblk; ++blk) {
        const int s0 = blk * 32;

        // K^T B-fragments: contiguous from row-major K (B col = key index)
        Frag kb[2][2];
        #pragma unroll
        for (int j = 0; j < 2; ++j) {
            const unsigned short* krow = Kbf + (b * Tt + s0 + j * 16 + lr) * Hh;
            #pragma unroll
            for (int k = 0; k < 2; ++k) {
                kb[j][k].u[0] = *reinterpret_cast<const uint4*>(krow + k * 32 + hf * 16);
                kb[j][k].u[1] = *reinterpret_cast<const uint4*>(krow + k * 32 + hf * 16 + 8);
            }
        }

        v8f sA = v8f{};
        sA = __builtin_amdgcn_wmma_f32_16x16x32_bf16(false, qa[0].v, false, kb[0][0].v, (short)0, sA, false, false);
        sA = __builtin_amdgcn_wmma_f32_16x16x32_bf16(false, qa[1].v, false, kb[0][1].v, (short)0, sA, false, false);
        v8f sB = v8f{};
        sB = __builtin_amdgcn_wmma_f32_16x16x32_bf16(false, qa[0].v, false, kb[1][0].v, (short)0, sB, false, false);
        sB = __builtin_amdgcn_wmma_f32_16x16x32_bf16(false, qa[1].v, false, kb[1][1].v, (short)0, sB, false, false);

        // Online softmax, per row r held by this lane (row = r + hf*8)
        float p0[8], p1[8];
        #pragma unroll
        for (int r = 0; r < 8; ++r) {
            const int row  = t0 + r + hf * 8;
            const int key0 = s0 + lr;
            const int key1 = s0 + 16 + lr;
            float v0 = (key0 <= row) ? sA[r] * scale : -1e30f;
            float v1 = (key1 <= row) ? sB[r] * scale : -1e30f;
            float mx = fmaxf(v0, v1);
            mx = fmaxf(mx, __shfl_xor(mx, 1, 32));
            mx = fmaxf(mx, __shfl_xor(mx, 2, 32));
            mx = fmaxf(mx, __shfl_xor(mx, 4, 32));
            mx = fmaxf(mx, __shfl_xor(mx, 8, 32));
            const float mnew = fmaxf(mi[r], mx);
            const float corr = __expf(mi[r] - mnew);
            v0 = __expf(v0 - mnew);
            v1 = __expf(v1 - mnew);
            float ps = v0 + v1;
            ps += __shfl_xor(ps, 1, 32);
            ps += __shfl_xor(ps, 2, 32);
            ps += __shfl_xor(ps, 4, 32);
            ps += __shfl_xor(ps, 8, 32);
            li[r] = li[r] * corr + ps;
            mi[r] = mnew;
            p0[r] = v0;
            p1[r] = v1;
            o[0][r] *= corr; o[1][r] *= corr; o[2][r] *= corr; o[3][r] *= corr;
        }

        // P: D layout -> A layout via LDS (16x32 bf16 tile)
        __syncthreads();   // WAR vs previous iteration's reads
        #pragma unroll
        for (int r = 0; r < 8; ++r) {
            Ptile[(r + hf * 8) * 32 + lr]      = bf_bits(p0[r]);
            Ptile[(r + hf * 8) * 32 + 16 + lr] = bf_bits(p1[r]);
        }
        __syncthreads();   // emits s_wait_dscnt + barrier
        Frag pa;
        pa.u[0] = *reinterpret_cast<const uint4*>(&Ptile[lr * 32 + hf * 16]);
        pa.u[1] = *reinterpret_cast<const uint4*>(&Ptile[lr * 32 + hf * 16 + 8]);

        // O += P V : V^T layout makes B-fragments contiguous (col = head dim)
        #pragma unroll
        for (int ng = 0; ng < 4; ++ng) {
            const unsigned short* vrow = Vt + (b * Hh + ng * 16 + lr) * Tt + s0 + hf * 16;
            Frag vb;
            vb.u[0] = *reinterpret_cast<const uint4*>(vrow);
            vb.u[1] = *reinterpret_cast<const uint4*>(vrow + 8);
            o[ng] = __builtin_amdgcn_wmma_f32_16x16x32_bf16(
                false, pa.v, false, vb.v, (short)0, o[ng], false, false);
        }
    }

    // Normalize and store fp32 output [B,T,H]
    #pragma unroll
    for (int r = 0; r < 8; ++r) {
        const float inv = 1.0f / li[r];
        const int row = t0 + r + hf * 8;
        #pragma unroll
        for (int ng = 0; ng < 4; ++ng)
            out[(b * Tt + row) * Hh + ng * 16 + lr] = o[ng][r] * inv;
    }
}

// ---------------------------------------------------------------------------
extern "C" void kernel_launch(void* const* d_in, const int* in_sizes, int n_in,
                              void* d_out, int out_size, void* d_ws, size_t ws_size,
                              hipStream_t stream) {
    const float* X  = (const float*)d_in[0];
    const float* Wk = (const float*)d_in[1];
    const float* Wq = (const float*)d_in[2];
    const float* Wv = (const float*)d_in[3];
    float* out = (float*)d_out;

    char* ws = (char*)d_ws;
    unsigned short* Wp  = (unsigned short*)ws;                       // 384 KB packed weights
    unsigned short* Qbf = (unsigned short*)(ws + 393216);            // 2 MB
    unsigned short* Kbf = (unsigned short*)(ws + 393216 + 2097152);  // 2 MB
    unsigned short* Vt  = (unsigned short*)(ws + 393216 + 4194304);  // 2 MB (transposed)

    pack_w_kernel<<<768, 256, 0, stream>>>(Wq, Wk, Wv, Wp);
    proj_kernel<<<256, 128, 0, stream>>>(X, Wp, Qbf, Kbf, Vt);
    flash_kernel<<<Bb * (Tt / 16), 32, 0, stream>>>(Qbf, Kbf, Vt, out);
}
